// BigBirdRegressor_MLP_cat_42150809043591
// MI455X (gfx1250) — matrix operand order repaired and checked by reading, not compiled, not run on hardware
//
#include <hip/hip_runtime.h>
#include <math.h>

// ---------------------------------------------------------------------------
// Types for CDNA5 WMMA (wave32)
// ---------------------------------------------------------------------------
typedef __attribute__((ext_vector_type(16))) __bf16 v16bf;
typedef __attribute__((ext_vector_type(8)))  __bf16 v8bf;
typedef __attribute__((ext_vector_type(8)))  float  v8f;

#define BB   4
#define SS   4096
#define DD   256
#define HH   4
#define DHH  64
#define LL   4
#define FFN  1024
#define NBLK 64
#define MTOT (BB * SS)   // 16384 rows

__device__ __forceinline__ float gelu_new(float x) {
    float x3 = x * x * x;
    return 0.5f * x * (1.0f + tanhf(0.7978845608028654f * (x + 0.044715f * x3)));
}

// ---------------------------------------------------------------------------
// CDNA5 async global->LDS copy (ASYNCcnt-tracked), 16B per lane.
// LDS byte address comes from the vdst VGPR; global addr = 64-bit VGPR pair (GV mode).
// ---------------------------------------------------------------------------
__device__ __forceinline__ void async_copy16(const __bf16* g, __bf16* l) {
    unsigned lds_off = (unsigned)(size_t)l;                 // low 32 bits = LDS byte offset
    unsigned long long ga = (unsigned long long)(size_t)g;  // flat == global address
    asm volatile("global_load_async_to_lds_b128 %0, %1, off"
                 :: "v"(lds_off), "v"(ga) : "memory");
}
__device__ __forceinline__ void async_wait0() {
    asm volatile("s_wait_asynccnt 0" ::: "memory");
}

// ---------------------------------------------------------------------------
// Weight convert + transpose: W[K,N] fp32 -> out[N,K] bf16 (LDS tile transpose)
// grid = (K/32, N/32), block = 256
// ---------------------------------------------------------------------------
__global__ __launch_bounds__(256) void cvt_t_kernel(const float* __restrict__ W,
                                                    __bf16* __restrict__ out,
                                                    int Kd, int Nd) {
    __shared__ float tile[32][33];
    int kb = blockIdx.x * 32, nb = blockIdx.y * 32;
    int tx = threadIdx.x & 31, ty = threadIdx.x >> 5;
#pragma unroll
    for (int i = 0; i < 32; i += 8)
        tile[ty + i][tx] = W[(size_t)(kb + ty + i) * Nd + nb + tx];
    __syncthreads();
#pragma unroll
    for (int i = 0; i < 32; i += 8)
        out[(size_t)(nb + ty + i) * Kd + kb + tx] = (__bf16)tile[tx][ty + i];
}

// ---------------------------------------------------------------------------
// Embedding gather: of[row][c] = word_emb[id][c] + pos_emb[s][c] + tok_emb[0][c]
// ---------------------------------------------------------------------------
__global__ __launch_bounds__(256) void embed_kernel(const int* __restrict__ ids,
                                                    const float* __restrict__ we,
                                                    const float* __restrict__ pe,
                                                    const float* __restrict__ te,
                                                    float* __restrict__ of) {
    int row = blockIdx.x;            // b * S + s
    int s   = row % SS;
    int t   = threadIdx.x;
    int id  = ids[row];
    of[(size_t)row * DD + t] = we[id * DD + t] + pe[(size_t)s * DD + t] + te[t];
}

// ---------------------------------------------------------------------------
// LayerNorm over D=256 of (a [+ b]); writes fp32 and bf16 mirrors.
// ---------------------------------------------------------------------------
__global__ __launch_bounds__(256) void ln_kernel(const float* __restrict__ a,
                                                 const float* __restrict__ b,
                                                 const float* __restrict__ g,
                                                 const float* __restrict__ beta,
                                                 float* __restrict__ outf,
                                                 __bf16* __restrict__ outh) {
    __shared__ float red[256];
    int row = blockIdx.x;
    int t   = threadIdx.x;
    size_t idx = (size_t)row * DD + t;
    float v = a[idx] + (b ? b[idx] : 0.0f);

    red[t] = v; __syncthreads();
    for (int o = 128; o > 0; o >>= 1) { if (t < o) red[t] += red[t + o]; __syncthreads(); }
    float mean = red[0] * (1.0f / DD);
    __syncthreads();
    float d = v - mean;
    red[t] = d * d; __syncthreads();
    for (int o = 128; o > 0; o >>= 1) { if (t < o) red[t] += red[t + o]; __syncthreads(); }
    float var = red[0] * (1.0f / DD);

    float y = d * rsqrtf(var + 1e-12f) * g[t] + beta[t];
    outf[idx] = y;
    outh[idx] = (__bf16)y;
}

// ---------------------------------------------------------------------------
// WMMA GEMM: C[M,N] = A[M,K](bf16 row-major) * Bt[N,K]^T(bf16 N-major) + bias
// BM=128, BN=64, BK=64. 8 waves, each a 32x32 macro tile (2x2 WMMA tiles).
// Double-buffered: both A and B tiles staged with async global->LDS B128
// copies; next stage is issued while WMMAs consume the current one.
// ACT: 0=none, 1=gelu_new.  OUTB/OUTF: write bf16 / f32 outputs.
// ---------------------------------------------------------------------------
#define GBM 128
#define GBN 64
#define GBK 64
#define ASTR 72   // 144 B row stride (multiple of 16 B)
#define BSTR 72

template <int ACT, int OUTB, int OUTF>
__global__ __launch_bounds__(256) void gemm_bf16_kernel(const __bf16* __restrict__ A,
                                                        const __bf16* __restrict__ Bt,
                                                        const float* __restrict__ bias,
                                                        __bf16* __restrict__ outb,
                                                        float* __restrict__ outf,
                                                        int M, int N, int K) {
    __shared__ __bf16 Asm[2][GBM][ASTR];
    __shared__ __bf16 Btm[2][GBN][BSTR];

    const int t    = threadIdx.x;
    const int w    = t >> 5;
    const int lane = t & 31;
    const int lo16 = lane & 15;
    const int hi   = lane >> 4;

    const int m0 = blockIdx.x * GBM;
    const int n0 = blockIdx.y * GBN;
    const int wm = (w & 3) * 32;   // wave row base within block tile
    const int wn = (w >> 2) * 32;  // wave col base within block tile

    v8f acc[2][2] = {};

    auto stage = [&](int buf, int k0) {
        // A: 128x64 bf16 = 1024 x 16B segs, 4 per thread
#pragma unroll
        for (int i = 0; i < 4; ++i) {
            int seg = t + i * 256;
            int r = seg >> 3, c = (seg & 7) * 8;
            async_copy16(&A[(size_t)(m0 + r) * K + k0 + c], &Asm[buf][r][c]);
        }
        // Bt: 64x64 bf16 = 512 x 16B segs, 2 per thread
#pragma unroll
        for (int i = 0; i < 2; ++i) {
            int seg = t + i * 256;
            int r = seg >> 3, c = (seg & 7) * 8;
            async_copy16(&Bt[(size_t)(n0 + r) * K + k0 + c], &Btm[buf][r][c]);
        }
    };

    stage(0, 0);
    int buf = 0;
    for (int k0 = 0; k0 < K; k0 += GBK, buf ^= 1) {
        async_wait0();
        __syncthreads();
        if (k0 + GBK < K) stage(buf ^ 1, k0 + GBK);   // prefetch next tile (async)

#pragma unroll
        for (int kk = 0; kk < GBK; kk += 32) {
            v16bf af[2], bfg[2];
#pragma unroll
            for (int i = 0; i < 2; ++i) {
                int m = wm + i * 16 + lo16;
                v8bf a0 = *(const v8bf*)&Asm[buf][m][kk + 8 * hi];
                v8bf a1 = *(const v8bf*)&Asm[buf][m][kk + 16 + 8 * hi];
#pragma unroll
                for (int e = 0; e < 8; ++e) { af[i][e] = a0[e]; af[i][e + 8] = a1[e]; }
                int n = wn + i * 16 + lo16;
                v8bf b0 = *(const v8bf*)&Btm[buf][n][kk + 8 * hi];
                v8bf b1 = *(const v8bf*)&Btm[buf][n][kk + 16 + 8 * hi];
#pragma unroll
                for (int e = 0; e < 8; ++e) { bfg[i][e] = b0[e]; bfg[i][e + 8] = b1[e]; }
            }
#pragma unroll
            for (int i = 0; i < 2; ++i)
#pragma unroll
                for (int j = 0; j < 2; ++j)
                    acc[i][j] = __builtin_amdgcn_wmma_f32_16x16x32_bf16(
                        false, af[i], false, bfg[j], (short)0, acc[i][j], false, false);
        }
    }

    // epilogue: bias (+gelu), write bf16 / f32
#pragma unroll
    for (int i = 0; i < 2; ++i) {
#pragma unroll
        for (int j = 0; j < 2; ++j) {
            int colg = n0 + wn + j * 16 + lo16;
            float bval = bias ? bias[colg] : 0.0f;
#pragma unroll
            for (int r = 0; r < 8; ++r) {
                int rowg = m0 + wm + i * 16 + r + 8 * hi;
                float v = acc[i][j][r] + bval;
                if (ACT == 1) v = gelu_new(v);
                if (OUTF) outf[(size_t)rowg * N + colg] = v;
                if (OUTB) outb[(size_t)rowg * N + colg] = (__bf16)v;
            }
        }
    }
}

// ---------------------------------------------------------------------------
// V transpose: vh[b*S+s, h*64+dh] -> vt[((b*H+h)*64+dh)*S + s]
// grid = (S/64, H, B), block = 256.
// ---------------------------------------------------------------------------
__global__ __launch_bounds__(256) void transpose_v_kernel(const __bf16* __restrict__ vh,
                                                          __bf16* __restrict__ vt) {
    __shared__ __bf16 tl[64][72];
    const int sb = blockIdx.x, h = blockIdx.y, b = blockIdx.z;
    const int t = threadIdx.x;
#pragma unroll
    for (int i = 0; i < 2; ++i) {
        int seg = t + i * 256;
        int r = seg >> 3;        // s within block
        int c = (seg & 7) * 8;   // dh
        *(v8bf*)&tl[r][c] = *(const v8bf*)&vh[((size_t)b * SS + sb * 64 + r) * DD + h * DHH + c];
    }
    __syncthreads();
#pragma unroll
    for (int i = 0; i < 2; ++i) {
        int seg = t + i * 256;
        int r = seg >> 3;        // dh
        int c = (seg & 7) * 8;   // s within block
        v8bf o;
#pragma unroll
        for (int e = 0; e < 8; ++e) o[e] = tl[c + e][r];
        *(v8bf*)&vt[(((size_t)b * HH + h) * DHH + r) * SS + sb * 64 + c] = o;
    }
}

// ---------------------------------------------------------------------------
// BigBird block-sparse attention, flash-style streaming softmax, bf16 WMMA.
// One workgroup per (query block qb, head h, batch b). 8 waves.
// Q/K/V tiles staged with async global->LDS B128 copies (V pre-transposed).
// ---------------------------------------------------------------------------
__global__ __launch_bounds__(256) void bigbird_attn_kernel(const __bf16* __restrict__ qh,
                                                           const __bf16* __restrict__ kh,
                                                           const __bf16* __restrict__ vt,
                                                           const int* __restrict__ attend,
                                                           __bf16* __restrict__ oh) {
    __shared__ __bf16 Qs[64][72];
    __shared__ __bf16 Ks[64][72];
    __shared__ __bf16 Vst[64][72];   // [dh][key]
    __shared__ __bf16 Ps[64][72];
    __shared__ float  Sf[64][66];
    __shared__ float  mrow[64], lrow[64], arow[64];

    const int qb = blockIdx.x, h = blockIdx.y, b = blockIdx.z;
    const int t = threadIdx.x, w = t >> 5, lane = t & 31;
    const int lo16 = lane & 15, hi = lane >> 4;
    const int tm = w >> 1;            // 0..3: 16-row strip
    const int cn = (w & 1) * 32;      // 32-col strip (2 tiles)
    const float scale = 0.125f;       // 1/sqrt(64)

    const size_t rowbase = (size_t)b * SS + qb * 64;
    const size_t vtbase  = ((size_t)b * HH + h) * DHH;   // row base in vt

    // stage Q (64x64 bf16) via async copies
#pragma unroll
    for (int i = 0; i < 2; ++i) {
        int seg = t + i * 256;
        int r = seg >> 3;
        int c = (seg & 7) * 8;
        async_copy16(&qh[(rowbase + r) * DD + h * DHH + c], &Qs[r][c]);
    }
    if (t < 64) { mrow[t] = -3.0e38f; lrow[t] = 0.0f; }

    v8f accO[2] = {};
    const bool edge = (qb == 0) || (qb == NBLK - 1);
    const int nkb = edge ? NBLK : 8;
    const int* lst = edge ? nullptr : (attend + (qb - 1) * 8);

    for (int jj = 0; jj < nkb; ++jj) {
        int kb = edge ? jj : lst[jj];
        // stage K (natural) and V (pre-transposed) via async copies
#pragma unroll
        for (int i = 0; i < 2; ++i) {
            int seg = t + i * 256;
            int r = seg >> 3;
            int c = (seg & 7) * 8;
            async_copy16(&kh[((size_t)b * SS + kb * 64 + r) * DD + h * DHH + c], &Ks[r][c]);
            async_copy16(&vt[(vtbase + r) * SS + kb * 64 + c], &Vst[r][c]);
        }
        async_wait0();
        __syncthreads();

        // S = Q * K^T (64x64), each wave: 2 tiles, K-loop over dh
        v8f accS[2] = {};
#pragma unroll
        for (int kk = 0; kk < 64; kk += 32) {
            v16bf aq;
            {
                int m = tm * 16 + lo16;
                v8bf a0 = *(const v8bf*)&Qs[m][kk + 8 * hi];
                v8bf a1 = *(const v8bf*)&Qs[m][kk + 16 + 8 * hi];
#pragma unroll
                for (int e = 0; e < 8; ++e) { aq[e] = a0[e]; aq[e + 8] = a1[e]; }
            }
#pragma unroll
            for (int j = 0; j < 2; ++j) {
                int n = cn + j * 16 + lo16;   // key index
                v16bf bk;
                v8bf b0 = *(const v8bf*)&Ks[n][kk + 8 * hi];
                v8bf b1 = *(const v8bf*)&Ks[n][kk + 16 + 8 * hi];
#pragma unroll
                for (int e = 0; e < 8; ++e) { bk[e] = b0[e]; bk[e + 8] = b1[e]; }
                accS[j] = __builtin_amdgcn_wmma_f32_16x16x32_bf16(
                    false, aq, false, bk, (short)0, accS[j], false, false);
            }
        }
#pragma unroll
        for (int j = 0; j < 2; ++j)
#pragma unroll
            for (int r = 0; r < 8; ++r)
                Sf[tm * 16 + r + 8 * hi][cn + j * 16 + lo16] = accS[j][r] * scale;
        __syncthreads();

        // online softmax: thread r owns row r
        if (t < 64) {
            float mo = mrow[t];
            float cm = -3.0e38f;
            for (int c2 = 0; c2 < 64; ++c2) cm = fmaxf(cm, Sf[t][c2]);
            float mn = fmaxf(mo, cm);
            float al = __expf(mo - mn);
            float ls = 0.0f;
            for (int c2 = 0; c2 < 64; ++c2) {
                float p = __expf(Sf[t][c2] - mn);
                Ps[t][c2] = (__bf16)p;
                ls += p;
            }
            mrow[t] = mn;
            lrow[t] = lrow[t] * al + ls;
            arow[t] = al;
        }
        __syncthreads();

        // rescale O accumulators, then O += P * V
#pragma unroll
        for (int j = 0; j < 2; ++j)
#pragma unroll
            for (int r = 0; r < 8; ++r)
                accO[j][r] *= arow[tm * 16 + r + 8 * hi];
#pragma unroll
        for (int kk = 0; kk < 64; kk += 32) {
            v16bf ap;
            {
                int m = tm * 16 + lo16;
                v8bf a0 = *(const v8bf*)&Ps[m][kk + 8 * hi];
                v8bf a1 = *(const v8bf*)&Ps[m][kk + 16 + 8 * hi];
#pragma unroll
                for (int e = 0; e < 8; ++e) { ap[e] = a0[e]; ap[e + 8] = a1[e]; }
            }
#pragma unroll
            for (int j = 0; j < 2; ++j) {
                int n = cn + j * 16 + lo16;   // dh index
                v16bf bv;
                v8bf b0 = *(const v8bf*)&Vst[n][kk + 8 * hi];
                v8bf b1 = *(const v8bf*)&Vst[n][kk + 16 + 8 * hi];
#pragma unroll
                for (int e = 0; e < 8; ++e) { bv[e] = b0[e]; bv[e + 8] = b1[e]; }
                accO[j] = __builtin_amdgcn_wmma_f32_16x16x32_bf16(
                    false, ap, false, bv, (short)0, accO[j], false, false);
            }
        }
        __syncthreads();   // protect Ks/Vst/Ps before next iteration's staging
    }

    if (t < 64) arow[t] = 1.0f / lrow[t];
    __syncthreads();
#pragma unroll
    for (int j = 0; j < 2; ++j) {
        int col = h * DHH + cn + j * 16 + lo16;
#pragma unroll
        for (int r = 0; r < 8; ++r) {
            int row = tm * 16 + r + 8 * hi;
            oh[(rowbase + row) * DD + col] = (__bf16)(accO[j][r] * arow[row]);
        }
    }
}

// ---------------------------------------------------------------------------
// Mean over hidden dim (D=256) -> pooled[B*S]. Wave per row.
// ---------------------------------------------------------------------------
__global__ __launch_bounds__(256) void pool_kernel(const float* __restrict__ xf,
                                                   float* __restrict__ pooled, int rows) {
    int row  = blockIdx.x * 8 + (threadIdx.x >> 5);
    int lane = threadIdx.x & 31;
    if (row >= rows) return;
    float s = 0.0f;
    for (int c = lane; c < DD; c += 32) s += xf[(size_t)row * DD + c];
    for (int o = 16; o > 0; o >>= 1) s += __shfl_xor(s, o, 32);
    if (lane == 0) pooled[row] = s * (1.0f / DD);
}

// ---------------------------------------------------------------------------
// Small GEMV head: out[b,n] = act(in[b,:] . W[:,n] + bias[n])
// grid = (N, B), block = 256.
// ---------------------------------------------------------------------------
__global__ __launch_bounds__(256) void fc_kernel(const float* __restrict__ in,
                                                 const float* __restrict__ W,
                                                 const float* __restrict__ bias,
                                                 float* __restrict__ out,
                                                 int K, int N, int relu) {
    __shared__ float red[256];
    int n = blockIdx.x, b = blockIdx.y, t = threadIdx.x;
    float s = 0.0f;
    for (int k = t; k < K; k += 256) s += in[(size_t)b * K + k] * W[(size_t)k * N + n];
    red[t] = s; __syncthreads();
    for (int o = 128; o > 0; o >>= 1) { if (t < o) red[t] += red[t + o]; __syncthreads(); }
    if (t == 0) {
        float v = red[0] + bias[n];
        if (relu) v = fmaxf(v, 0.0f);
        out[(size_t)b * N + n] = v;
    }
}

// ---------------------------------------------------------------------------
// Host launch
// ---------------------------------------------------------------------------
extern "C" void kernel_launch(void* const* d_in, const int* in_sizes, int n_in,
                              void* d_out, int out_size, void* d_ws, size_t ws_size,
                              hipStream_t stream) {
    (void)in_sizes; (void)n_in; (void)out_size; (void)ws_size;

    const int*   ids    = (const int*)d_in[0];
    const int*   attend = (const int*)d_in[1];
    const float* we     = (const float*)d_in[2];
    const float* pe     = (const float*)d_in[3];
    const float* te     = (const float*)d_in[4];
    const float* elns   = (const float*)d_in[5];
    const float* elnb   = (const float*)d_in[6];
    const float* Wq     = (const float*)d_in[7];
    const float* bq     = (const float*)d_in[8];
    const float* Wk     = (const float*)d_in[9];
    const float* bk     = (const float*)d_in[10];
    const float* Wv     = (const float*)d_in[11];
    const float* bv     = (const float*)d_in[12];
    const float* Wo     = (const float*)d_in[13];
    const float* bo     = (const float*)d_in[14];
    const float* ln1s   = (const float*)d_in[15];
    const float* ln1b   = (const float*)d_in[16];
    const float* Wi     = (const float*)d_in[17];
    const float* bi     = (const float*)d_in[18];
    const float* Wo2    = (const float*)d_in[19];
    const float* bo2    = (const float*)d_in[20];
    const float* ln2s   = (const float*)d_in[21];
    const float* ln2b   = (const float*)d_in[22];
    const float* fcW1   = (const float*)d_in[23];
    const float* fcb1   = (const float*)d_in[24];
    const float* fcW2   = (const float*)d_in[25];
    const float* fcb2   = (const float*)d_in[26];
    const float* fcW3   = (const float*)d_in[27];
    const float* fcb3   = (const float*)d_in[28];

    // workspace carve
    char* ws = (char*)d_ws;
    auto take = [&](size_t bytes) { char* p = ws; ws += (bytes + 255) & ~(size_t)255; return p; };
    float*  xf     = (float*)take((size_t)MTOT * DD * 4);
    __bf16* xh     = (__bf16*)take((size_t)MTOT * DD * 2);
    __bf16* qh     = (__bf16*)take((size_t)MTOT * DD * 2);
    __bf16* kh     = (__bf16*)take((size_t)MTOT * DD * 2);
    __bf16* vh     = (__bf16*)take((size_t)MTOT * DD * 2);
    __bf16* vtp    = (__bf16*)take((size_t)MTOT * DD * 2);    // V transposed [b,h,dh,s]
    __bf16* ah     = (__bf16*)take((size_t)MTOT * DD * 2);
    float*  of     = (float*)take((size_t)MTOT * DD * 4);
    __bf16* ffh    = (__bf16*)take((size_t)MTOT * FFN * 2);
    __bf16* wb     = (__bf16*)take((size_t)FFN * DD * 2);     // transposed-weight scratch
    float*  pooled = (float*)take((size_t)MTOT * 4);
    float*  h1     = (float*)take((size_t)BB * 1000 * 4);
    float*  h2     = (float*)take((size_t)BB * 500 * 4);

    const dim3 blk(256);
    const dim3 gemmDD(MTOT / GBM, DD / GBN);    // (128, 4)
    const dim3 gemmFF(MTOT / GBM, FFN / GBN);   // (128, 16)
    auto cvtT = [&](const float* src, __bf16* dst, int Kd, int Nd) {
        cvt_t_kernel<<<dim3(Kd / 32, Nd / 32), blk, 0, stream>>>(src, dst, Kd, Nd);
    };

    // embedding + LN
    embed_kernel<<<MTOT, blk, 0, stream>>>(ids, we, pe, te, of);
    ln_kernel<<<MTOT, blk, 0, stream>>>(of, nullptr, elns, elnb, xf, xh);

    for (int l = 0; l < LL; ++l) {
        const float* Wql  = Wq  + (size_t)l * DD * DD;
        const float* Wkl  = Wk  + (size_t)l * DD * DD;
        const float* Wvl  = Wv  + (size_t)l * DD * DD;
        const float* Wol  = Wo  + (size_t)l * DD * DD;
        const float* Wil  = Wi  + (size_t)l * DD * FFN;
        const float* Wo2l = Wo2 + (size_t)l * FFN * DD;

        // Q/K/V projections (bf16 out)
        cvtT(Wql, wb, DD, DD);
        gemm_bf16_kernel<0, 1, 0><<<gemmDD, blk, 0, stream>>>(xh, wb, bq + l * DD, qh, nullptr, MTOT, DD, DD);
        cvtT(Wkl, wb, DD, DD);
        gemm_bf16_kernel<0, 1, 0><<<gemmDD, blk, 0, stream>>>(xh, wb, bk + l * DD, kh, nullptr, MTOT, DD, DD);
        cvtT(Wvl, wb, DD, DD);
        gemm_bf16_kernel<0, 1, 0><<<gemmDD, blk, 0, stream>>>(xh, wb, bv + l * DD, vh, nullptr, MTOT, DD, DD);

        // transpose V for the attention B-operand, then block-sparse attention
        transpose_v_kernel<<<dim3(SS / 64, HH, BB), blk, 0, stream>>>(vh, vtp);
        bigbird_attn_kernel<<<dim3(NBLK, HH, BB), blk, 0, stream>>>(qh, kh, vtp, attend, ah);

        // output projection (f32 out) + residual LN
        cvtT(Wol, wb, DD, DD);
        gemm_bf16_kernel<0, 0, 1><<<gemmDD, blk, 0, stream>>>(ah, wb, bo + l * DD, nullptr, of, MTOT, DD, DD);
        ln_kernel<<<MTOT, blk, 0, stream>>>(xf, of, ln1s + l * DD, ln1b + l * DD, xf, xh);

        // FFN: gelu(x@Wi+bi) @ Wo2 + bo2, residual LN
        cvtT(Wil, wb, DD, FFN);
        gemm_bf16_kernel<1, 1, 0><<<gemmFF, blk, 0, stream>>>(xh, wb, bi + l * FFN, ffh, nullptr, MTOT, FFN, DD);
        cvtT(Wo2l, wb, FFN, DD);
        gemm_bf16_kernel<0, 0, 1><<<gemmDD, blk, 0, stream>>>(ffh, wb, bo2 + l * DD, nullptr, of, MTOT, DD, FFN);
        ln_kernel<<<MTOT, blk, 0, stream>>>(xf, of, ln2s + l * DD, ln2b + l * DD, xf, xh);
    }

    // pool over hidden dim, then MLP head
    pool_kernel<<<MTOT / 8, blk, 0, stream>>>(xf, pooled, MTOT);
    fc_kernel<<<dim3(1000, BB), blk, 0, stream>>>(pooled, fcW1, fcb1, h1, SS, 1000, 1);
    fc_kernel<<<dim3(500, BB), blk, 0, stream>>>(h1, fcW2, fcb2, h2, 1000, 500, 1);
    fc_kernel<<<dim3(1, BB), blk, 0, stream>>>(h2, fcW3, fcb3, (float*)d_out, 500, 1, 0);
}
